// Criterion_47029891891453
// MI455X (gfx1250) — matrix operand order repaired and checked by reading, not compiled, not run on hardware
//
#include <hip/hip_runtime.h>
#include <hip/hip_bf16.h>

// ---------------------------------------------------------------------------
// Margin triplet loss, CDNA5 (gfx1250, wave32) version.
//   B=4096, D=1024, T=65536, NCLS=100, MARGIN=0.2, EPS=1e-8
//
// d_ap^2 = ||a||^2 + ||p||^2 - 2 a.p ; cross terms via v_wmma_f32_16x16x32_bf16
// (16 triplets per wave, diagonal of the 16x16 accumulator), norms in fp32.
// ---------------------------------------------------------------------------

#define BDIM   4096
#define DDIM   1024
#define TTRIP  65536
#define MARGIN 0.2f
#define EPSV   1e-8f

typedef __attribute__((ext_vector_type(16))) __bf16 v16bf;
typedef __attribute__((ext_vector_type(8)))  __bf16 v8bf;
typedef __attribute__((ext_vector_type(4)))  __bf16 v4bf;
typedef __attribute__((ext_vector_type(8)))  float  v8f;

union FragA {            // 16x32 bf16 A-slice, 8 VGPRs per lane
    v16bf f;
    v8bf  h[2];
};
union FragB {            // 32x16 bf16 B-slice, 8 VGPRs per lane
    v16bf f;
};

// Select v[i] with an unrolled cndmask chain (i is per-lane dynamic, 0..7).
__device__ __forceinline__ float vsel8(v8f v, int i) {
    float r = v[0];
#pragma unroll
    for (int k = 1; k < 8; ++k) r = (i == k) ? v[k] : r;
    return r;
}

// ---------------------------------------------------------------------------
// Kernel 1: bf16 copy of batch + fp32 row norms + zero accumulators.
// One block per row, 256 threads, float4 per thread.
// ---------------------------------------------------------------------------
__global__ void __launch_bounds__(256)
prep_kernel(const float* __restrict__ batch,
            __bf16* __restrict__ bbf,
            float* __restrict__ norms,
            float* __restrict__ accum) {
    const int row = blockIdx.x;
    const int t   = threadIdx.x;

    const float4* src = (const float4*)(batch + (size_t)row * DDIM);
    v4bf*         dst = (v4bf*)(bbf + (size_t)row * DDIM);

    float4 v = src[t];                       // 1024 / 256 = 4 floats/thread
    float  s = v.x * v.x + v.y * v.y + v.z * v.z + v.w * v.w;

    v4bf o;
    o[0] = (__bf16)v.x; o[1] = (__bf16)v.y;
    o[2] = (__bf16)v.z; o[3] = (__bf16)v.w;
    dst[t] = o;

    __shared__ float red[256];
    red[t] = s;
    __syncthreads();
#pragma unroll
    for (int off = 128; off > 0; off >>= 1) {
        if (t < off) red[t] += red[t + off];
        __syncthreads();
    }
    if (t == 0) norms[row] = red[0];
    if (row == 0 && t < 2) accum[t] = 0.0f;  // [0]=total, [1]=pair_count
}

// ---------------------------------------------------------------------------
// Kernel 2: one wave (32 lanes) per 16-triplet tile.
//   A  = 16 anchor rows       (16 x 1024, row-major)
//   Bp = 16 positive rows as columns (1024 x 16)
//   Bn = 16 negative rows as columns (1024 x 16)
// Accumulate diag(A Bp) and diag(A Bn) over 32 K-slices of 32.
// ---------------------------------------------------------------------------
__global__ void __launch_bounds__(256)
triplet_wmma_kernel(const __bf16* __restrict__ bbf,
                    const float*  __restrict__ norms,
                    const int*    __restrict__ labels,
                    const int*    __restrict__ trips,
                    const float*  __restrict__ beta,
                    float*        __restrict__ accum) {
    const int wave = blockIdx.x * 8 + (threadIdx.x >> 5);   // tile id, 0..4095
    const int lane = threadIdx.x & 31;
    const int hi   = lane >> 4;                             // 0: lanes 0-15, 1: 16-31

    // Triplet handled by this lane's matrix slot (column / row index = lane & 15).
    const int j   = wave * 16 + (lane & 15);
    const int ia  = trips[3 * j + 0];
    const int ip  = trips[3 * j + 1];
    const int in_ = trips[3 * j + 2];

    const __bf16* arow = bbf + (size_t)ia  * DDIM;
    const __bf16* prow = bbf + (size_t)ip  * DDIM;
    const __bf16* nrow = bbf + (size_t)in_ * DDIM;

    // Per-lane base offsets; this code uses contiguous k order, the same for A and B:
    //  A 16x32 bf16: lane<16 -> K 0..15 contiguous; lane>=16 -> K 16..31
    //  B 32x16 bf16: lane<16 -> K 0..15 contiguous; lane>=16 -> K 16..31
    const int aoff = hi * 16;
    const int boff = hi * 16;

    v8f cap = {};   // diag accumulators live inside full 16x16 C tiles
    v8f can = {};

#pragma unroll 4
    for (int ks = 0; ks < 32; ++ks) {
        const int k0 = ks * 32;

        FragA au;
        au.h[0] = *(const v8bf*)(arow + k0 + aoff);
        au.h[1] = *(const v8bf*)(arow + k0 + aoff + 8);

        FragB pu, nu;
        pu.f = *(const v16bf*)(prow + k0 + boff);
        nu.f = *(const v16bf*)(nrow + k0 + boff);

        cap = __builtin_amdgcn_wmma_f32_16x16x32_bf16(
                  false, au.f, false, pu.f, (short)0, cap, false, false);
        can = __builtin_amdgcn_wmma_f32_16x16x32_bf16(
                  false, au.f, false, nu.f, (short)0, can, false, false);
    }

    // Diagonal extraction (16x16 f32 C/D layout):
    //   (j,j), j<8  -> lane j,      VGPR j
    //   (j,j), j>=8 -> lane j+16,   VGPR j-8   => lanes 24..31, VGPR lane-24
    float psum = 0.0f, pcnt = 0.0f;
    const bool act = (lane < 8) || (lane >= 24);
    if (act) {
        const int vi = (lane < 8) ? lane : (lane - 24);
        const float dot_ap = vsel8(cap, vi);
        const float dot_an = vsel8(can, vi);

        const float na  = norms[ia];
        const float npo = norms[ip];
        const float nn  = norms[in_];
        const float bt  = beta[labels[ia]];

        const float d2ap = fmaxf(fmaf(-2.0f, dot_ap, na + npo), 0.0f);
        const float d2an = fmaxf(fmaf(-2.0f, dot_an, na + nn),  0.0f);
        const float dap  = sqrtf(d2ap + EPSV);
        const float dan  = sqrtf(d2an + EPSV);

        const float pl = fmaxf(dap - bt + MARGIN, 0.0f);
        const float nl = fmaxf(bt - dan + MARGIN, 0.0f);
        psum = pl + nl;
        pcnt = (pl > 0.0f ? 1.0f : 0.0f) + (nl > 0.0f ? 1.0f : 0.0f);
    }

    // wave32 reduction, then two device-scope f32 atomics per wave.
#pragma unroll
    for (int off = 16; off > 0; off >>= 1) {
        psum += __shfl_down(psum, off, 32);
        pcnt += __shfl_down(pcnt, off, 32);
    }
    if (lane == 0) {
        atomicAdd(&accum[0], psum);
        atomicAdd(&accum[1], pcnt);
    }
}

// ---------------------------------------------------------------------------
// Kernel 3: finalize scalar loss.
// ---------------------------------------------------------------------------
__global__ void finalize_kernel(const float* __restrict__ accum,
                                float* __restrict__ out) {
    const float total = accum[0];
    const float cnt   = accum[1];
    out[0] = (cnt == 0.0f) ? total : (total / cnt);
}

// ---------------------------------------------------------------------------
extern "C" void kernel_launch(void* const* d_in, const int* in_sizes, int n_in,
                              void* d_out, int out_size, void* d_ws, size_t ws_size,
                              hipStream_t stream) {
    const float* batch  = (const float*)d_in[0];   // [4096,1024] f32
    const int*   labels = (const int*)  d_in[1];   // [4096]
    const int*   trips  = (const int*)  d_in[2];   // [65536,3]
    const float* beta   = (const float*)d_in[3];   // [101]
    float*       out    = (float*)d_out;

    char*   ws    = (char*)d_ws;
    __bf16* bbf   = (__bf16*)ws;                                   // 8 MB
    float*  norms = (float*)(ws + (size_t)BDIM * DDIM * sizeof(__bf16));
    float*  accum = norms + BDIM;                                  // 2 floats

    prep_kernel<<<BDIM, 256, 0, stream>>>(batch, bbf, norms, accum);
    triplet_wmma_kernel<<<TTRIP / 16 / 8, 256, 0, stream>>>(
        bbf, norms, labels, trips, beta, accum);
    finalize_kernel<<<1, 1, 0, stream>>>(accum, out);
}
